// Block_37091337568777
// MI455X (gfx1250) — matrix operand order repaired and checked by reading, not compiled, not run on hardware
//
#include <hip/hip_runtime.h>

typedef __bf16 bf16_t;
typedef __bf16 v16bf __attribute__((ext_vector_type(16)));
typedef float  v8f   __attribute__((ext_vector_type(8)));

#define NX     768
#define NHEAD  12
#define SCTX   2048
#define NBATCH 4
#define NTOK   (NBATCH * SCTX)   // 8192 tokens
#define DH     64
#define NEGV   (-1000000000.0f)

// ---------------------------------------------------------------------------
// WMMA helpers (CDNA5 wave32, 16x16x32 bf16 -> f32)
// ---------------------------------------------------------------------------
static __device__ __forceinline__ v8f wmma_bf16(v16bf a, v16bf b, v8f c) {
  return __builtin_amdgcn_wmma_f32_16x16x32_bf16(false, a, false, b, (short)0, c,
                                                 false, false);
}

static __device__ __forceinline__ v8f zero8() {
  v8f z = {0.f, 0.f, 0.f, 0.f, 0.f, 0.f, 0.f, 0.f};
  return z;
}

// A-matrix fragment (16x32 bf16, row-major source, lda in elements).
// ISA layout: lane L holds m=L%16; elems 0..7 -> k=(L/16)*8+0..7, elems 8..15 -> +16.
static __device__ __forceinline__ v16bf load_a_frag(const bf16_t* p0, int lda, int lane) {
  int m = lane & 15, half = lane >> 4;
  const bf16_t* p = p0 + (size_t)m * lda + half * 8;
  union { v16bf v; uint4 u[2]; } f;
  f.u[0] = *(const uint4*)(p);
  f.u[1] = *(const uint4*)(p + 16);
  return f.v;
}

// B-matrix fragment (32x16 bf16). Source is B^T row-major [N][K] (ldb = K).
// ISA layout: lane L holds n=L%16; elems 0..7 -> k=(L/16)*8+0..7, elems 8..15 -> +16.
static __device__ __forceinline__ v16bf load_b_frag(const bf16_t* p0, int ldb, int lane) {
  int n = lane & 15, half = lane >> 4;
  const bf16_t* p = p0 + (size_t)n * ldb + half * 8;
  union { v16bf v; uint4 u[2]; } f;
  f.u[0] = *(const uint4*)(p);
  f.u[1] = *(const uint4*)(p + 16);
  return f.v;
}

// ---------------------------------------------------------------------------
// Elementwise / repack kernels
// ---------------------------------------------------------------------------
__global__ void cvt_f32_bf16(const float* __restrict__ x, bf16_t* __restrict__ y, int n) {
  int i = blockIdx.x * blockDim.x + threadIdx.x;
  if (i < n) y[i] = (bf16_t)x[i];
}

// w[K][N] f32 (row-major) -> wt[N][K] bf16 (i.e., column-major / B^T)
__global__ void wtrans(const float* __restrict__ w, bf16_t* __restrict__ wt, int K, int N) {
  int i = blockIdx.x * blockDim.x + threadIdx.x;
  if (i < K * N) {
    int k = i / N, n = i - k * N;
    wt[(size_t)n * K + k] = (bf16_t)w[i];
  }
}

// qkv f32 [NTOK][3*NX] -> qh/kh bf16 [B*H][S][dh], vt bf16 [B*H][dh][S]
__global__ void split_heads(const float* __restrict__ qkv, bf16_t* __restrict__ qh,
                            bf16_t* __restrict__ kh, bf16_t* __restrict__ vt) {
  int i = blockIdx.x * blockDim.x + threadIdx.x;
  if (i >= NTOK * NX) return;
  int col = i % NX;
  int row = i / NX;
  int h = col / DH, d = col - h * DH;
  int b = row / SCTX, s = row - b * SCTX;
  int bh = b * NHEAD + h;
  const float* src = qkv + (size_t)row * (3 * NX);
  qh[((size_t)bh * SCTX + s) * DH + d] = (bf16_t)src[col];
  kh[((size_t)bh * SCTX + s) * DH + d] = (bf16_t)src[NX + col];
  vt[((size_t)bh * DH + d) * SCTX + s] = (bf16_t)src[2 * NX + col];
}

// ---------------------------------------------------------------------------
// Generic bf16 WMMA GEMM: C[M][N] = A[M][K] * B[K][N] + bias, B given as B^T.
// Block = 4 waves; each wave computes a 64x64 tile (4x4 of 16x16): 16 WMMAs
// per K=32 step against 8 fragment loads -> 1:1 wmma : b128-load issue ratio.
// ---------------------------------------------------------------------------
__global__ __launch_bounds__(128) void gemm_bf16_kernel(
    const bf16_t* __restrict__ A, const bf16_t* __restrict__ Bt,
    const float* __restrict__ bias, float* __restrict__ Cf, bf16_t* __restrict__ Cb,
    int M, int N, int K, int relu) {
  const int lane = threadIdx.x & 31, wave = threadIdx.x >> 5;
  const int tileM = blockIdx.y * 128 + (wave >> 1) * 64;
  const int tileN = blockIdx.x * 128 + (wave & 1) * 64;

  v8f acc[4][4];
#pragma unroll
  for (int tm = 0; tm < 4; ++tm)
#pragma unroll
    for (int tn = 0; tn < 4; ++tn) acc[tm][tn] = zero8();

  const bf16_t* ap = A + (size_t)tileM * K;
  const bf16_t* bp = Bt + (size_t)tileN * K;

  for (int k = 0; k < K; k += 32) {
    v16bf af[4], bf[4];
#pragma unroll
    for (int i = 0; i < 4; ++i)
      af[i] = load_a_frag(ap + (size_t)(16 * i) * K + k, K, lane);
#pragma unroll
    for (int i = 0; i < 4; ++i)
      bf[i] = load_b_frag(bp + (size_t)(16 * i) * K + k, K, lane);
#pragma unroll
    for (int tm = 0; tm < 4; ++tm)
#pragma unroll
      for (int tn = 0; tn < 4; ++tn)
        acc[tm][tn] = wmma_bf16(af[tm], bf[tn], acc[tm][tn]);
  }

  const int n16 = lane & 15, halfm = lane >> 4;
#pragma unroll
  for (int tm = 0; tm < 4; ++tm)
#pragma unroll
    for (int tn = 0; tn < 4; ++tn) {
      int col = tileN + tn * 16 + n16;
      float bv = bias[col];
#pragma unroll
      for (int j = 0; j < 8; ++j) {
        float v = acc[tm][tn][j] + bv;
        if (relu) v = fmaxf(v, 0.f);
        int row = tileM + tm * 16 + halfm * 8 + j;
        size_t idx = (size_t)row * N + col;
        if (Cf) Cf[idx] = v;
        if (Cb) Cb[idx] = (bf16_t)v;
      }
    }
}

// ---------------------------------------------------------------------------
// Fused causal attention (no 1/sqrt(d) scale, per reference scale=False).
// One block (8 waves) per (batch*head, 16-query-row tile). Each wave owns a
// 256-wide strip of keys: WMMA scores -> softmax (shfl + LDS reduce) ->
// WMMA P*V over its own strip -> ds_add_f32 combine.
// ---------------------------------------------------------------------------
__global__ __launch_bounds__(256) void attn_kernel(
    const bf16_t* __restrict__ qh, const bf16_t* __restrict__ kh,
    const bf16_t* __restrict__ vt, bf16_t* __restrict__ ab) {
  __shared__ float red[2][8][16];
  __shared__ float oacc[16 * DH];
  __shared__ __align__(16) bf16_t tbuf[8][16 * 32];

  const int tid = threadIdx.x;
  const int wave = tid >> 5, lane = tid & 31;
  const int n16 = lane & 15, halfm = lane >> 4;

  const int qt = blockIdx.x & 127;      // S/16 = 128 query tiles
  const int bh = blockIdx.x >> 7;
  const int b = bh / NHEAD, h = bh - b * NHEAD;
  const int qbase = qt * 16;

  const bf16_t* Q  = qh + ((size_t)bh * SCTX + qbase) * DH;
  const bf16_t* Kt = kh + (size_t)bh * SCTX * DH;   // [S][dh] == B^T for q@k^T
  const bf16_t* Vt = vt + (size_t)bh * DH * SCTX;   // [dh][S] == B^T for p@v

  for (int i = tid; i < 16 * DH; i += 256) oacc[i] = 0.f;

  v16bf aq0 = load_a_frag(Q, DH, lane);       // k = 0..31
  v16bf aq1 = load_a_frag(Q + 32, DH, lane);  // k = 32..63

  const int strip = wave * 256;
  v8f accs[16];
#pragma unroll
  for (int t = 0; t < 16; ++t) {
    int nbase = strip + t * 16;
    v8f c = zero8();
    if (nbase <= qbase + 15) {     // not fully masked (wave-uniform branch)
      v16bf b0 = load_b_frag(Kt + (size_t)nbase * DH, DH, lane);
      v16bf b1 = load_b_frag(Kt + (size_t)nbase * DH + 32, DH, lane);
      c = wmma_bf16(aq0, b0, c);
      c = wmma_bf16(aq1, b1, c);
      if (nbase + 15 > qbase) {    // straddles diagonal: elementwise mask
        int key = nbase + n16;
#pragma unroll
        for (int j = 0; j < 8; ++j)
          if (key > qbase + halfm * 8 + j) c[j] = NEGV;
      }
    } else {
#pragma unroll
      for (int j = 0; j < 8; ++j) c[j] = NEGV;
    }
    accs[t] = c;
  }

  // ---- row max (16 lanes hold one row per (half,j)) ----
  float rmax[8];
#pragma unroll
  for (int j = 0; j < 8; ++j) rmax[j] = -3.0e38f;
#pragma unroll
  for (int t = 0; t < 16; ++t)
#pragma unroll
    for (int j = 0; j < 8; ++j) rmax[j] = fmaxf(rmax[j], accs[t][j]);
#pragma unroll
  for (int m = 1; m < 16; m <<= 1)
#pragma unroll
    for (int j = 0; j < 8; ++j) rmax[j] = fmaxf(rmax[j], __shfl_xor(rmax[j], m, 32));
  if (n16 == 0)
#pragma unroll
    for (int j = 0; j < 8; ++j) red[0][wave][halfm * 8 + j] = rmax[j];
  __syncthreads();
  float gmax[8];
#pragma unroll
  for (int j = 0; j < 8; ++j) {
    float g = -3.0e38f;
    for (int w = 0; w < 8; ++w) g = fmaxf(g, red[0][w][halfm * 8 + j]);
    gmax[j] = g;
  }

  // ---- exp + row sum ----
  float rsum[8];
#pragma unroll
  for (int j = 0; j < 8; ++j) rsum[j] = 0.f;
#pragma unroll
  for (int t = 0; t < 16; ++t)
#pragma unroll
    for (int j = 0; j < 8; ++j) {
      float e = __expf(accs[t][j] - gmax[j]);
      accs[t][j] = e;
      rsum[j] += e;
    }
#pragma unroll
  for (int m = 1; m < 16; m <<= 1)
#pragma unroll
    for (int j = 0; j < 8; ++j) rsum[j] += __shfl_xor(rsum[j], m, 32);
  if (n16 == 0)
#pragma unroll
    for (int j = 0; j < 8; ++j) red[1][wave][halfm * 8 + j] = rsum[j];
  __syncthreads();
  float rinv[8];
#pragma unroll
  for (int j = 0; j < 8; ++j) {
    float ssum = 0.f;
    for (int w = 0; w < 8; ++w) ssum += red[1][w][halfm * 8 + j];
    rinv[j] = 1.f / ssum;
  }

  // ---- O_partial = P_strip @ V_strip (C-layout -> A-layout via per-wave LDS) ----
  v8f oa[4];
#pragma unroll
  for (int t = 0; t < 4; ++t) oa[t] = zero8();
  bf16_t* myb = tbuf[wave];
#pragma unroll 1
  for (int ks = 0; ks < 8; ++ks) {
#pragma unroll
    for (int tt = 0; tt < 2; ++tt) {
      int t = ks * 2 + tt;
#pragma unroll
      for (int j = 0; j < 8; ++j) {
        int m = halfm * 8 + j;
        myb[m * 32 + tt * 16 + n16] = (bf16_t)(accs[t][j] * rinv[j]);
      }
    }
    v16bf ap = load_a_frag(myb, 32, lane);        // per-wave LDS, in-order DS pipe
    const bf16_t* vb = Vt + strip + ks * 32;
#pragma unroll
    for (int nt = 0; nt < 4; ++nt) {
      v16bf bv = load_b_frag(vb + (size_t)nt * 16 * SCTX, SCTX, lane);
      oa[nt] = wmma_bf16(ap, bv, oa[nt]);
    }
  }

  __syncthreads();
#pragma unroll
  for (int nt = 0; nt < 4; ++nt)
#pragma unroll
    for (int j = 0; j < 8; ++j) {
      int m = halfm * 8 + j, d = nt * 16 + n16;
      atomicAdd(&oacc[m * DH + d], oa[nt][j]);    // ds_add_f32
    }
  __syncthreads();
  for (int i = tid; i < 16 * DH; i += 256) {
    int m = i >> 6, d = i & 63;
    ab[((size_t)(b * SCTX + qbase + m)) * NX + h * DH + d] = (bf16_t)oacc[i];
  }
}

// ---------------------------------------------------------------------------
// LayerNorm with the reference's additive bug: out = g + (t-u)/sqrt(s+eps) + b
// where t = r1 + r2. One block per token row (768 = 3*256).
// ---------------------------------------------------------------------------
__global__ __launch_bounds__(256) void ln_kernel(
    const float* __restrict__ r1, const float* __restrict__ r2,
    const float* __restrict__ g, const float* __restrict__ bb,
    float* __restrict__ of, bf16_t* __restrict__ ob) {
  __shared__ float sred[256];
  const int tid = threadIdx.x;
  const size_t base = (size_t)blockIdx.x * NX;
  float t[3];
  float s = 0.f;
#pragma unroll
  for (int i = 0; i < 3; ++i) {
    int idx = tid + i * 256;
    t[i] = r1[base + idx] + r2[base + idx];
    s += t[i];
  }
  sred[tid] = s;
  __syncthreads();
  for (int st = 128; st > 0; st >>= 1) {
    if (tid < st) sred[tid] += sred[tid + st];
    __syncthreads();
  }
  float mean = sred[0] * (1.f / NX);
  __syncthreads();
  float v = 0.f;
#pragma unroll
  for (int i = 0; i < 3; ++i) {
    float d = t[i] - mean;
    v += d * d;
  }
  sred[tid] = v;
  __syncthreads();
  for (int st = 128; st > 0; st >>= 1) {
    if (tid < st) sred[tid] += sred[tid + st];
    __syncthreads();
  }
  float inv = rsqrtf(sred[0] * (1.f / NX) + 1e-5f);
#pragma unroll
  for (int i = 0; i < 3; ++i) {
    int idx = tid + i * 256;
    float o = g[idx] + (t[i] - mean) * inv + bb[idx];
    of[base + idx] = o;
    if (ob) ob[base + idx] = (bf16_t)o;
  }
}

// ---------------------------------------------------------------------------
// Orchestration
// ---------------------------------------------------------------------------
extern "C" void kernel_launch(void* const* d_in, const int* in_sizes, int n_in,
                              void* d_out, int out_size, void* d_ws, size_t ws_size,
                              hipStream_t stream) {
  const float* x      = (const float*)d_in[0];
  const float* w_attn = (const float*)d_in[1];
  const float* b_attn = (const float*)d_in[2];
  const float* w_proj = (const float*)d_in[3];
  const float* b_proj = (const float*)d_in[4];
  const float* ln1_g  = (const float*)d_in[5];
  const float* ln1_b  = (const float*)d_in[6];
  const float* w_fc   = (const float*)d_in[7];
  const float* b_fc   = (const float*)d_in[8];
  const float* w_fc2  = (const float*)d_in[9];
  const float* b_fc2  = (const float*)d_in[10];
  const float* ln2_g  = (const float*)d_in[11];
  const float* ln2_b  = (const float*)d_in[12];
  (void)in_sizes; (void)n_in; (void)out_size; (void)ws_size;

  char* ws = (char*)d_ws;
  size_t off = 0;
  auto alloc = [&](size_t bytes) -> char* {
    char* p = ws + off;
    off += (bytes + 255) & ~(size_t)255;
    return p;
  };

  bf16_t* xb   = (bf16_t*)alloc((size_t)NTOK * NX * 2);        // x in bf16
  bf16_t* wat  = (bf16_t*)alloc((size_t)3 * NX * NX * 2);      // w_attn^T bf16
  bf16_t* wpt  = (bf16_t*)alloc((size_t)NX * NX * 2);          // w_proj^T bf16
  bf16_t* wft  = (bf16_t*)alloc((size_t)4 * NX * NX * 2);      // w_fc^T bf16
  bf16_t* wf2t = (bf16_t*)alloc((size_t)4 * NX * NX * 2);      // w_fc2^T bf16
  float*  qkv  = (float*) alloc((size_t)NTOK * 3 * NX * 4);    // qkv f32
  bf16_t* qh   = (bf16_t*)alloc((size_t)NTOK * NX * 2);
  bf16_t* kh   = (bf16_t*)alloc((size_t)NTOK * NX * 2);
  bf16_t* vtb  = (bf16_t*)alloc((size_t)NTOK * NX * 2);
  bf16_t* ab   = (bf16_t*)alloc((size_t)NTOK * NX * 2);        // attn out bf16
  float*  aF   = (float*) alloc((size_t)NTOK * NX * 4);        // proj out f32
  float*  nF   = (float*) alloc((size_t)NTOK * NX * 4);        // ln1 out f32
  bf16_t* nb   = (bf16_t*)alloc((size_t)NTOK * NX * 2);        // ln1 out bf16
  bf16_t* hb   = (bf16_t*)alloc((size_t)NTOK * 4 * NX * 2);    // relu(fc) bf16
  float*  mF   = (float*) alloc((size_t)NTOK * NX * 4);        // fc2 out f32

  // precision conversion / weight repack
  cvt_f32_bf16<<<(NTOK * NX + 255) / 256, 256, 0, stream>>>(x, xb, NTOK * NX);
  wtrans<<<(NX * 3 * NX + 255) / 256, 256, 0, stream>>>(w_attn, wat, NX, 3 * NX);
  wtrans<<<(NX * NX + 255) / 256, 256, 0, stream>>>(w_proj, wpt, NX, NX);
  wtrans<<<(NX * 4 * NX + 255) / 256, 256, 0, stream>>>(w_fc, wft, NX, 4 * NX);
  wtrans<<<(4 * NX * NX + 255) / 256, 256, 0, stream>>>(w_fc2, wf2t, 4 * NX, NX);

  // qkv = x @ w_attn + b_attn
  gemm_bf16_kernel<<<dim3(3 * NX / 128, NTOK / 128), 128, 0, stream>>>(
      xb, wat, b_attn, qkv, nullptr, NTOK, 3 * NX, NX, 0);
  split_heads<<<(NTOK * NX + 255) / 256, 256, 0, stream>>>(qkv, qh, kh, vtb);

  // causal attention
  attn_kernel<<<NBATCH * NHEAD * (SCTX / 16), 256, 0, stream>>>(qh, kh, vtb, ab);

  // a = attn @ w_proj + b_proj ;  n = LN1(x + a)
  gemm_bf16_kernel<<<dim3(NX / 128, NTOK / 128), 128, 0, stream>>>(
      ab, wpt, b_proj, aF, nullptr, NTOK, NX, NX, 0);
  ln_kernel<<<NTOK, 256, 0, stream>>>(x, aF, ln1_g, ln1_b, nF, nb);

  // h = relu(n @ w_fc + b_fc)  (bf16 out, fused)
  gemm_bf16_kernel<<<dim3(4 * NX / 128, NTOK / 128), 128, 0, stream>>>(
      nb, wft, b_fc, nullptr, hb, NTOK, 4 * NX, NX, 1);
  // m = h @ w_fc2 + b_fc2
  gemm_bf16_kernel<<<dim3(NX / 128, NTOK / 128), 128, 0, stream>>>(
      hb, wf2t, b_fc2, mF, nullptr, NTOK, NX, 4 * NX, 0);

  // out = LN2(n + m)
  ln_kernel<<<NTOK, 256, 0, stream>>>(nF, mF, ln2_g, ln2_b, (float*)d_out, nullptr);
}
